// GATLayer_6828998001538
// MI455X (gfx1250) — matrix operand order repaired and checked by reading, not compiled, not run on hardware
//
#include <hip/hip_runtime.h>

typedef __attribute__((ext_vector_type(16))) _Float16 v16h;
typedef __attribute__((ext_vector_type(8)))  float    v8f;

#define IN_DIM  64
#define OUT_DIM 64

// Sign-correct float atomic max via int/uint ordering trick.
__device__ __forceinline__ void atomicMaxF(float* addr, float val) {
    if (val >= 0.0f)
        atomicMax((int*)addr, __float_as_int(val));
    else
        atomicMin((unsigned int*)addr, __float_as_uint(val));
}

// ---------------------------------------------------------------------------
// K0: init emax=-inf, denom=0, out=0
// ---------------------------------------------------------------------------
__global__ void k_init(float* __restrict__ emax, float* __restrict__ denom,
                       float* __restrict__ out, int n_nodes) {
    int i = blockIdx.x * blockDim.x + threadIdx.x;
    if (i < n_nodes) {
        emax[i]  = -__builtin_inff();
        denom[i] = 0.0f;
    }
    if (i < n_nodes * OUT_DIM) out[i] = 0.0f;
}

// ---------------------------------------------------------------------------
// K1: P = H * M^T via v_wmma_f32_16x16x32_f16.
//   M[o,i] = w.flat[o*64+i]  =>  B[k][n] = w[n*64+k] (contiguous in k per lane)
//   One wave -> 16 rows x 64 cols of P, 8 waves/block -> 128 rows/block.
// ---------------------------------------------------------------------------
__global__ void k_gemm_wmma(const float* __restrict__ h, const float* __restrict__ w,
                            float* __restrict__ p, int n_nodes) {
    const int lane = threadIdx.x & 31;
    const int wv   = threadIdx.x >> 5;
    const int lr   = lane & 15;       // row within A-tile / col within B-tile
    const int hi   = lane >> 4;       // half-wave selector
    const int rowBase = blockIdx.x * 128 + wv * 16;

    int row = rowBase + lr;
    if (row >= n_nodes) row = n_nodes - 1;   // clamp loads; stores predicated

    v8f acc[4];
#pragma unroll
    for (int nt = 0; nt < 4; ++nt)
        acc[nt] = (v8f){0.f, 0.f, 0.f, 0.f, 0.f, 0.f, 0.f, 0.f};

#pragma unroll
    for (int ks = 0; ks < IN_DIM; ks += 32) {
        // A fragment: lanes<16 -> K=ks+{0..15}; lanes>=16 -> K=ks+{16..31}
        v16h a;
        const float* hp = h + (size_t)row * IN_DIM + ks + hi * 16;
#pragma unroll
        for (int j = 0; j < 8; ++j) {
            a[j]     = (_Float16)hp[j];
            a[8 + j] = (_Float16)hp[8 + j];
        }
        // B fragments: lane col n = nt*16+lr, K = ks + hi*16 + {0..15}
        const int kb = ks + hi * 16;
#pragma unroll
        for (int nt = 0; nt < 4; ++nt) {
            const float* wp = w + (size_t)(nt * 16 + lr) * IN_DIM + kb;
            v16h b;
#pragma unroll
            for (int j = 0; j < 16; ++j) b[j] = (_Float16)wp[j];
            acc[nt] = __builtin_amdgcn_wmma_f32_16x16x32_f16(
                false, a, false, b, (short)0, acc[nt], false, false);
        }
    }

    // C/D layout: VGPR v, lanes 0-15 -> M=v, lanes 16-31 -> M=v+8; N = lr
#pragma unroll
    for (int nt = 0; nt < 4; ++nt) {
#pragma unroll
        for (int v = 0; v < 8; ++v) {
            int r = rowBase + hi * 8 + v;
            if (r < n_nodes)
                p[(size_t)r * OUT_DIM + nt * 16 + lr] = acc[nt][v];
        }
    }
}

// ---------------------------------------------------------------------------
// K2: per-node attention projections  ss[n] = h[n]·a_s,  sd[n] = h[n]·a_d
// ---------------------------------------------------------------------------
__global__ void k_attn_proj(const float* __restrict__ h, const float* __restrict__ attn,
                            float* __restrict__ ss, float* __restrict__ sd, int n_nodes) {
    int i = blockIdx.x * blockDim.x + threadIdx.x;
    if (i >= n_nodes) return;
    const float4* hp = (const float4*)(h + (size_t)i * IN_DIM);
    const float4* as = (const float4*)attn;
    const float4* ad = (const float4*)(attn + IN_DIM);
    float a = 0.f, b = 0.f;
#pragma unroll
    for (int j = 0; j < IN_DIM / 4; ++j) {
        float4 v = hp[j], s = as[j], d = ad[j];
        a += v.x * s.x + v.y * s.y + v.z * s.z + v.w * s.w;
        b += v.x * d.x + v.y * d.y + v.z * d.z + v.w * d.w;
    }
    ss[i] = a;
    sd[i] = b;
}

// ---------------------------------------------------------------------------
// K3: edge logits e = leaky_relu(ss[src]+sd[dst]); running segment-max on dst
// ---------------------------------------------------------------------------
__global__ void k_edge_logits(const int* __restrict__ src, const int* __restrict__ dst,
                              const float* __restrict__ ss, const float* __restrict__ sd,
                              float* __restrict__ ebuf, float* __restrict__ emax, int ne) {
    int e = blockIdx.x * blockDim.x + threadIdx.x;
    if (e >= ne) return;
    float v = ss[src[e]] + sd[dst[e]];
    v = v > 0.f ? v : 0.01f * v;
    ebuf[e] = v;
    atomicMaxF(&emax[dst[e]], v);
}

// ---------------------------------------------------------------------------
// K4: ex = exp(e - emax[dst]); denom[dst] += ex
// ---------------------------------------------------------------------------
__global__ void k_edge_exp(const int* __restrict__ dst, const float* __restrict__ ebuf,
                           const float* __restrict__ emax, float* __restrict__ exbuf,
                           float* __restrict__ denom, int ne) {
    int e = blockIdx.x * blockDim.x + threadIdx.x;
    if (e >= ne) return;
    int d = dst[e];
    float ex = __expf(ebuf[e] - emax[d]);
    exbuf[e] = ex;
    atomicAdd(&denom[d], ex);
}

// ---------------------------------------------------------------------------
// K5: out[dst] += (ex/denom[dst]) * w_comp[we] * P[src]   (16 threads/edge, 4 f/thread)
// ---------------------------------------------------------------------------
__global__ void k_scatter(const int* __restrict__ src, const int* __restrict__ dst,
                          const int* __restrict__ we, const float* __restrict__ wcomp,
                          const float* __restrict__ p, const float* __restrict__ exbuf,
                          const float* __restrict__ denom, float* __restrict__ out, int ne) {
    long long t = (long long)blockIdx.x * blockDim.x + threadIdx.x;
    int e = (int)(t >> 4);
    if (e >= ne) return;
    int q = (int)(t & 15);
    int d = dst[e];
    float coef = exbuf[e] / fmaxf(denom[d], 1e-38f) * wcomp[we[e]];
    float4 pv = ((const float4*)(p + (size_t)src[e] * OUT_DIM))[q];
    float* op = out + (size_t)d * OUT_DIM + q * 4;
    atomicAdd(op + 0, pv.x * coef);
    atomicAdd(op + 1, pv.y * coef);
    atomicAdd(op + 2, pv.z * coef);
    atomicAdd(op + 3, pv.w * coef);
}

extern "C" void kernel_launch(void* const* d_in, const int* in_sizes, int n_in,
                              void* d_out, int out_size, void* d_ws, size_t ws_size,
                              hipStream_t stream) {
    (void)n_in; (void)out_size; (void)ws_size;
    const float* h      = (const float*)d_in[0];
    const int*   src    = (const int*)d_in[1];
    const int*   dst    = (const int*)d_in[2];
    const int*   we     = (const int*)d_in[3];
    const float* w      = (const float*)d_in[4];
    const float* w_comp = (const float*)d_in[5];
    const float* attn_w = (const float*)d_in[6];
    // d_in[7] = fc_w: dead code in reference (z never returned)
    float* out = (float*)d_out;

    const int n_nodes = in_sizes[0] / IN_DIM;
    const int n_edges = in_sizes[1];

    // workspace carve-out (256B aligned slabs)
    char* ws = (char*)d_ws;
    size_t off = 0;
    auto alloc = [&](size_t bytes) -> char* {
        char* r = ws + off;
        off += (bytes + 255) & ~(size_t)255;
        return r;
    };
    float* p     = (float*)alloc((size_t)n_nodes * OUT_DIM * sizeof(float));
    float* ss    = (float*)alloc((size_t)n_nodes * sizeof(float));
    float* sd    = (float*)alloc((size_t)n_nodes * sizeof(float));
    float* emax  = (float*)alloc((size_t)n_nodes * sizeof(float));
    float* denom = (float*)alloc((size_t)n_nodes * sizeof(float));
    float* ebuf  = (float*)alloc((size_t)n_edges * sizeof(float));
    float* exbuf = (float*)alloc((size_t)n_edges * sizeof(float));

    const int total = n_nodes * OUT_DIM;
    k_init<<<(total + 255) / 256, 256, 0, stream>>>(emax, denom, out, n_nodes);
    k_gemm_wmma<<<(n_nodes + 127) / 128, 256, 0, stream>>>(h, w, p, n_nodes);
    k_attn_proj<<<(n_nodes + 255) / 256, 256, 0, stream>>>(h, attn_w, ss, sd, n_nodes);
    k_edge_logits<<<(n_edges + 255) / 256, 256, 0, stream>>>(src, dst, ss, sd, ebuf, emax, n_edges);
    k_edge_exp<<<(n_edges + 255) / 256, 256, 0, stream>>>(dst, ebuf, emax, exbuf, denom, n_edges);
    const long long st = (long long)n_edges * 16;
    k_scatter<<<(int)((st + 255) / 256), 256, 0, stream>>>(src, dst, we, w_comp, p, exbuf,
                                                           denom, out, n_edges);
}
